// ConvCaps_9337258901594
// MI455X (gfx1250) — matrix operand order repaired and checked by reading, not compiled, not run on hardware
//
#include <hip/hip_runtime.h>
#include <stdint.h>

// ---------------- problem constants ----------------
#define AA     32
#define BBJ    32          // B capsules (softmax width)
#define KC     3
#define CC     16          // pose vector len
#define KKA    288         // 9*32
#define HH     14
#define WW     14
#define OWD    6
#define LL     36          // 6*6
#define NBATCH 16
#define NPATCH 576         // 16*36
#define BD     512         // B*D channels
#define NSTEP  144         // 288 capsules / 2 per K=32 WMMA

typedef __attribute__((ext_vector_type(16))) __bf16        v16bf;
typedef __attribute__((ext_vector_type(8)))  float          v8f;

static __device__ __forceinline__ unsigned short f32_to_bf16(float f) {
  union { float f; unsigned u; } v; v.f = f;
  unsigned r = v.u + 0x7FFFu + ((v.u >> 16) & 1u);   // round-to-nearest-even
  return (unsigned short)(r >> 16);
}

// ---------------------------------------------------------------------------
// Kernel 1: repack W1 [288][512][16] f32 -> bf16 in the exact per-lane WMMA
// B-operand order. Flat dword d decodes as: v = d&7 (VGPR), lane = (d>>3)&31,
// j = (d>>8)&31, s = d>>13 (K-step, n0 = 2s).
// B layout (16-bit, K=32): half h = lane>>4; VGPR v<4 -> n0, c = 8h+2(v&3)..+1;
// v>=4 -> n0+1, c = 8h+2(v&3)..+1 (matches A).  N = lane&15 = d.
// ---------------------------------------------------------------------------
__global__ __launch_bounds__(256) void prep_w1_kernel(
    const float* __restrict__ W1, uint32_t* __restrict__ w1b) {
  const uint32_t d = blockIdx.x * 256u + threadIdx.x;   // 1,179,648 dwords
  const uint32_t v = d & 7u, lane = (d >> 3) & 31u, j = (d >> 8) & 31u, s = d >> 13;
  const uint32_t n = 2u * s + (v >> 2);
  const uint32_t m = j * 16u + (lane & 15u);
  const uint32_t c = 8u * (lane >> 4) + 2u * (v & 3u);
  const float* src = W1 + ((size_t)n * BD + m) * CC + c;
  const uint32_t lo = f32_to_bf16(src[0]);
  const uint32_t hi = f32_to_bf16(src[1]);
  w1b[d] = lo | (hi << 16);
}

__global__ void zero_stats_kernel(float* __restrict__ s1, float* __restrict__ s2) {
  const int t = threadIdx.x;           // 512 threads
  s1[t] = 0.0f;
  s2[t] = 0.0f;
}

// ---------------------------------------------------------------------------
// Kernel 2: per-patch routing. 1 block / patch, 288 threads (thread t = capsule n).
// Produces: bf16 pose_u rows, bf16 coeffT[j][n] (transposed!), and a_out.
// ---------------------------------------------------------------------------
__global__ __launch_bounds__(288) void routing_kernel(
    const float* __restrict__ act, const float* __restrict__ pose,
    const float* __restrict__ W2, const float* __restrict__ b2,
    unsigned short* __restrict__ pu_ws, unsigned short* __restrict__ coeffT_ws,
    float* __restrict__ a_out) {
  __shared__ float pu[KKA][CC];        // 18 KB
  __shared__ float au[KKA];            // 1.1 KB
  __shared__ float ar[KKA][BBJ + 1];   // 38 KB, padded (33 coprime with 64 banks)
  __shared__ float arsum[BBJ];
  __shared__ float ausum;

  const int pid = blockIdx.x;
  const int b = pid / LL, l = pid % LL;
  const int i0 = (l / OWD) * 2, j0 = (l % OWD) * 2;
  const int t = threadIdx.x;

  // gather pose patch: channel ch = a*16 + c; LDS row n = kk*32 + a
  for (int ch = t; ch < AA * CC; ch += 288) {
    const int ai = ch >> 4, c = ch & 15;
    const float* pp = pose + ((size_t)(b * (AA * CC) + ch) * HH + i0) * WW + j0;
#pragma unroll
    for (int kk = 0; kk < 9; ++kk)
      pu[kk * AA + ai][c] = pp[(kk / KC) * WW + (kk % KC)];
  }
  // gather activation patch: n = kk*32 + a
  {
    const int ai = t & 31, kk = t >> 5;
    au[t] = act[((size_t)(b * AA + ai) * HH + i0 + kk / KC) * WW + j0 + (kk % KC)];
  }
  __syncthreads();

  const int n = t;
  // export bf16 pose_u row for the WMMA A operand
  {
    unsigned short* dst = pu_ws + ((size_t)pid * KKA + n) * CC;
#pragma unroll
    for (int c = 0; c < CC; ++c) dst[c] = f32_to_bf16(pu[n][c]);
  }
  // fp32 logits + softmax over j
  float lg[BBJ];
  const float* w2r = W2 + (size_t)n * BBJ * CC;
#pragma unroll
  for (int j = 0; j < BBJ; ++j) {
    float s = b2[n * BBJ + j];
#pragma unroll
    for (int c = 0; c < CC; ++c) s += w2r[j * CC + c] * pu[n][c];
    lg[j] = s;
  }
  float mx = lg[0];
#pragma unroll
  for (int j = 1; j < BBJ; ++j) mx = fmaxf(mx, lg[j]);
  float se = 0.0f;
#pragma unroll
  for (int j = 0; j < BBJ; ++j) { lg[j] = __expf(lg[j] - mx); se += lg[j]; }
  const float scale = au[n] / se;      // ar = a_u * softmax
#pragma unroll
  for (int j = 0; j < BBJ; ++j) ar[n][j] = lg[j] * scale;
  __syncthreads();

  if (t < BBJ) {                       // ar_sum over n, per j
    float s = 0.0f;
    for (int k = 0; k < KKA; ++k) s += ar[k][t];
    arsum[t] = s;
  } else if (t == BBJ) {               // sum of a_u
    float s = 0.0f;
    for (int k = 0; k < KKA; ++k) s += au[k];
    ausum = s;
  }
  __syncthreads();

  if (t < BBJ)                          // a_out[b][j][oh][ow]
    a_out[((size_t)b * BBJ + t) * LL + l] = arsum[t] / ausum;

  // coeff, transposed to [patch][j][n] and stored bf16: the GEMM lane then
  // reads (n0, n0+1) as one dword.
  unsigned short* cdst = coeffT_ws + (size_t)pid * BBJ * KKA + n;
#pragma unroll
  for (int j = 0; j < BBJ; ++j)
    cdst[j * KKA] = f32_to_bf16(ar[n][j] / arsum[j]);
}

// ---------------------------------------------------------------------------
// Kernel 3: WMMA contraction. Grid = 36 patch-tiles x 4 j-groups; 8 waves per
// block, each wave owns one j (16 output channels). 144 x v_wmma per wave.
// A fragment (16-bit 16x32): row M = lane&15 (= patch), lanes<16 take c 0..7,
// lanes>=16 take c 8..15; elements 0..7 = capsule n0, 8..15 = n0+1.
// coeff is folded into A with v_pk_mul_bf16 (inline asm, packed bf16 mul).
// ---------------------------------------------------------------------------
__global__ __launch_bounds__(256) void gemm_kernel(
    const unsigned short* __restrict__ w1b,
    const unsigned short* __restrict__ pu_ws,
    const unsigned short* __restrict__ coeffT_ws,
    float* __restrict__ poseo, float* __restrict__ chsum,
    float* __restrict__ chsumsq) {
  const int lane = threadIdx.x & 31;
  const int wave = threadIdx.x >> 5;
  const int pt = blockIdx.x % (NPATCH / 16);
  const int jg = blockIdx.x / (NPATCH / 16);
  const int j = jg * 8 + wave;
  const int p0 = pt * 16;
  const int row = lane & 15;
  const int hiL = lane >> 4;
  const int p = p0 + row;               // this lane's A-row patch

  const unsigned short* pubase = pu_ws + (size_t)p * KKA * CC + hiL * 8;
  const unsigned short* cfT = coeffT_ws + ((size_t)p * BBJ + j) * KKA;
  const unsigned short* bbase = w1b + (((size_t)j * 32 + lane) << 4); // +16384/step

  v8f acc = {};
#pragma unroll 4
  for (int s = 0; s < NSTEP; ++s) {
    // A inputs: pose_u[p][n0][c0..c0+7] and [n0+1][...], n0 = 2s (packed bf16)
    const uint4 pa0 = *(const uint4*)(pubase + s * 32);
    const uint4 pa1 = *(const uint4*)(pubase + s * 32 + 16);
    // coeff pair (cf[n0], cf[n0+1]) in one dword
    const uint32_t w = *(const uint32_t*)(cfT + 2 * s);
    const uint32_t lo2 = (w & 0xffffu) * 0x10001u;   // splat cf0 into both halves
    const uint32_t hi2 = (w >> 16) * 0x10001u;       // splat cf1
    // B input: one contiguous 32B fragment per lane (L2-resident)
    const v16bf bvec = *(const v16bf*)(bbase + (size_t)s * 16384);
    if (s + 4 < NSTEP)
      __builtin_prefetch((const void*)(bbase + (size_t)(s + 4) * 16384), 0, 3);

    const uint32_t p0d[4] = {pa0.x, pa0.y, pa0.z, pa0.w};
    const uint32_t p1d[4] = {pa1.x, pa1.y, pa1.z, pa1.w};
    union { uint32_t d[8]; v16bf v; } afrag;
#pragma unroll
    for (int i = 0; i < 4; ++i) {
      asm("v_pk_mul_bf16 %0, %1, %2" : "=v"(afrag.d[i])     : "v"(p0d[i]), "v"(lo2));
      asm("v_pk_mul_bf16 %0, %1, %2" : "=v"(afrag.d[i + 4]) : "v"(p1d[i]), "v"(hi2));
    }
    acc = __builtin_amdgcn_wmma_f32_16x16x32_bf16(false, afrag.v, false, bvec,
                                                  (short)0, acc, false, false);
  }

  // D layout: VGPR v -> row M = v + 8*hiL (patch), col N = lane&15 (d)
  const int m = j * 16 + row;           // fixed channel for this lane
  float s1 = 0.0f, s2 = 0.0f;
#pragma unroll
  for (int v = 0; v < 8; ++v) {
    const float x = acc[v];
    poseo[(size_t)(p0 + v + 8 * hiL) * BD + m] = x;
    s1 += x;
    s2 += x * x;
  }
  s1 += __shfl_xor(s1, 16);             // combine rows 0-7 with 8-15 (same m)
  s2 += __shfl_xor(s2, 16);
  if (hiL == 0) {
    atomicAdd(&chsum[m], s1);
    atomicAdd(&chsumsq[m], s2);
  }
}

// ---------------------------------------------------------------------------
// Kernel 4: BatchNorm (training-mode batch stats) + final layout [b,512,6,6]
// ---------------------------------------------------------------------------
__global__ __launch_bounds__(256) void bn_kernel(
    const float* __restrict__ poseo, const float* __restrict__ chsum,
    const float* __restrict__ chsumsq, const float* __restrict__ gamma,
    const float* __restrict__ beta, float* __restrict__ pose_out) {
  const int idx = blockIdx.x * 256 + threadIdx.x;   // 294912
  const int patch = idx >> 9, m = idx & 511;
  const int b = patch / LL, l = patch % LL;
  const float inv_n = 1.0f / (float)NPATCH;
  const float mean = chsum[m] * inv_n;
  const float var = chsumsq[m] * inv_n - mean * mean;
  const float x = (poseo[idx] - mean) * rsqrtf(var + 1e-5f);
  pose_out[((size_t)b * BD + m) * LL + l] = x * gamma[m] + beta[m];
}

// ---------------------------------------------------------------------------
extern "C" void kernel_launch(void* const* d_in, const int* in_sizes, int n_in,
                              void* d_out, int out_size, void* d_ws, size_t ws_size,
                              hipStream_t stream) {
  const float* a     = (const float*)d_in[0];
  const float* pose  = (const float*)d_in[1];
  const float* W1    = (const float*)d_in[2];
  const float* W2    = (const float*)d_in[3];
  const float* b2    = (const float*)d_in[4];
  const float* gamma = (const float*)d_in[5];
  const float* beta  = (const float*)d_in[6];

  char* ws = (char*)d_ws;
  uint32_t*       w1b    = (uint32_t*)(ws);                        //  4,718,592 B
  unsigned short* pu     = (unsigned short*)(ws + 4718592);        //  5,308,416 B
  unsigned short* coeffT = (unsigned short*)(ws + 10027008);       // 10,616,832 B
  float*          poseo  = (float*)(ws + 20643840);                //  1,179,648 B
  float*          chsum  = (float*)(ws + 21823488);                //      2,048 B
  float*          chsq   = chsum + 512;                            //      2,048 B

  float* aout    = (float*)d_out;                 // 16*32*36 = 18432
  float* poseout = aout + NBATCH * BBJ * LL;      // 16*512*36 = 294912

  prep_w1_kernel<<<4608, 256, 0, stream>>>(W1, w1b);
  zero_stats_kernel<<<1, 512, 0, stream>>>(chsum, chsq);
  routing_kernel<<<NPATCH, 288, 0, stream>>>(a, pose, W2, b2, pu, coeffT, aout);
  gemm_kernel<<<(NPATCH / 16) * 4, 256, 0, stream>>>(
      (const unsigned short*)w1b, pu, coeffT, poseo, chsum, chsq);
  bn_kernel<<<(NPATCH * BD) / 256, 256, 0, stream>>>(poseo, chsum, chsq,
                                                     gamma, beta, poseout);
}
